// GMM_Layer_29996051595916
// MI455X (gfx1250) — matrix operand order repaired and checked by reading, not compiled, not run on hardware
//
#include <hip/hip_runtime.h>
#include <math.h>

typedef __attribute__((ext_vector_type(16))) _Float16 v16h;
typedef __attribute__((ext_vector_type(8)))  _Float16 v8h;
typedef __attribute__((ext_vector_type(8)))  float    v8f;

#define C_IN       64
#define K_OUT      100
#define K_PAD      112
#define F_DIM      128      // 2*C_IN features: [x, x^2]
#define N_PIX      32768    // B*H*W
#define LDS_STRIDE 136      // padded row stride (halves) to kill bank conflicts
#define NTILES     7        // K_PAD / 16

// ---------------- prep: log-softmax(pis), build split-precision W, w0 -------
__global__ void gmm_prep_kernel(const float* __restrict__ mus,
                                const float* __restrict__ sigmas,
                                const float* __restrict__ pis,
                                _Float16* __restrict__ whi,
                                _Float16* __restrict__ wlo,
                                float* __restrict__ w0) {
  __shared__ float sOff;
  if (threadIdx.x == 0) {
    float m = pis[0];
    for (int k = 1; k < K_OUT; ++k) m = fmaxf(m, pis[k]);
    float s = 0.f;
    for (int k = 0; k < K_OUT; ++k) s += expf(pis[k] - m);
    sOff = m + logf(s);
  }
  __syncthreads();
  for (int k = threadIdx.x; k < K_PAD; k += blockDim.x) {
    if (k < K_OUT) {
      float logdet = 0.f, quad = 0.f;
      for (int c = 0; c < C_IN; ++c) {
        float sg = sigmas[k * C_IN + c];
        float mu = mus[k * C_IN + c];
        float s2 = sg * sg;
        float w1 = mu * s2;       // weight for feature x_c
        float w2 = -0.5f * s2;    // weight for feature x_c^2
        _Float16 h1 = (_Float16)w1;
        _Float16 h2 = (_Float16)w2;
        whi[k * F_DIM + c]        = h1;
        wlo[k * F_DIM + c]        = (_Float16)(w1 - (float)h1);
        whi[k * F_DIM + C_IN + c] = h2;
        wlo[k * F_DIM + C_IN + c] = (_Float16)(w2 - (float)h2);
        logdet += logf(sg);
        quad   += mu * mu * s2;
      }
      // -0.5*64*log(2*pi) = -58.81206612509905
      w0[k] = -58.8120661251f + logdet - 0.5f * quad + (pis[k] - sOff);
    } else {
      for (int j = 0; j < F_DIM; ++j) {
        whi[k * F_DIM + j] = (_Float16)0.f;
        wlo[k * F_DIM + j] = (_Float16)0.f;
      }
      w0[k] = 0.f;
    }
  }
}

// ---------------- main: split-precision f16 WMMA GEMM -----------------------
__global__ __launch_bounds__(256)
void gmm_wmma_kernel(const float* __restrict__ x,
                     const _Float16* __restrict__ whi,
                     const _Float16* __restrict__ wlo,
                     const float* __restrict__ w0,
                     float* __restrict__ out) {
  __shared__ _Float16 sWhi[K_PAD * LDS_STRIDE];
  __shared__ _Float16 sWlo[K_PAD * LDS_STRIDE];

  // Stage W into LDS, restriding 128 -> 136 halves (dword copies).
  for (int i = threadIdx.x; i < K_PAD * (F_DIM / 2); i += blockDim.x) {
    int k  = i / (F_DIM / 2);
    int c2 = i % (F_DIM / 2);
    unsigned vh = ((const unsigned*)(whi + (size_t)k * F_DIM))[c2];
    unsigned vl = ((const unsigned*)(wlo + (size_t)k * F_DIM))[c2];
    *(unsigned*)&sWhi[k * LDS_STRIDE + 2 * c2] = vh;
    *(unsigned*)&sWlo[k * LDS_STRIDE + 2 * c2] = vl;
  }
  __syncthreads();

  const int lane  = threadIdx.x & 31;
  const int wave  = threadIdx.x >> 5;
  const int half  = lane >> 4;     // which half of the wave (K sub-block)
  const int l16   = lane & 15;
  const int mtile = blockIdx.x * 8 + wave;
  const int row0  = mtile * 16;
  const float* xr = x + (size_t)(row0 + l16) * C_IN;   // A-matrix row M = lane&15

  v8f acc[NTILES] = {};

  for (int kb = 0; kb < F_DIM; kb += 32) {
    // ---- build A fragment (ISA 16-bit A layout: two 8-wide K runs/lane) ----
    float av[16];
    {
      int f0 = kb + half * 8;                    // K = f0 + j, j=0..7
      const float4* p0 = (const float4*)(xr + (f0 & (C_IN - 1)));
      float4 q0 = p0[0], q1 = p0[1];
      float t0[8] = {q0.x, q0.y, q0.z, q0.w, q1.x, q1.y, q1.z, q1.w};
      bool sq0 = (f0 >= C_IN);
      #pragma unroll
      for (int j = 0; j < 8; ++j) av[j] = sq0 ? t0[j] * t0[j] : t0[j];

      int f1 = kb + 16 + half * 8;               // K = f1 + j, j=0..7
      const float4* p1 = (const float4*)(xr + (f1 & (C_IN - 1)));
      float4 q2 = p1[0], q3 = p1[1];
      float t1[8] = {q2.x, q2.y, q2.z, q2.w, q3.x, q3.y, q3.z, q3.w};
      bool sq1 = (f1 >= C_IN);
      #pragma unroll
      for (int j = 0; j < 8; ++j) av[8 + j] = sq1 ? t1[j] * t1[j] : t1[j];
    }
    v16h ahi, alo;
    #pragma unroll
    for (int j = 0; j < 16; ++j) {
      _Float16 h = (_Float16)av[j];
      ahi[j] = h;
      alo[j] = (_Float16)(av[j] - (float)h);
    }

    // ---- 7 N-tiles: B fragment from LDS (two 8-wide K runs/lane, same as A) ----
    #pragma unroll
    for (int nt = 0; nt < NTILES; ++nt) {
      int krow = nt * 16 + l16;                  // B column N = lane&15
      const _Float16* bhp = &sWhi[krow * LDS_STRIDE + kb + half * 8];
      const _Float16* blp = &sWlo[krow * LDS_STRIDE + kb + half * 8];
      v8h bh0 = *(const v8h*)(bhp);
      v8h bh1 = *(const v8h*)(bhp + 16);
      v8h bl0 = *(const v8h*)(blp);
      v8h bl1 = *(const v8h*)(blp + 16);
      v16h bhi, blo;
      #pragma unroll
      for (int j = 0; j < 8; ++j) {
        bhi[j] = bh0[j]; bhi[j + 8] = bh1[j];
        blo[j] = bl0[j]; blo[j + 8] = bl1[j];
      }
      // f16x3 split product: hi*hi + lo*hi + hi*lo (fp32 accumulate)
      acc[nt] = __builtin_amdgcn_wmma_f32_16x16x32_f16(
          false, ahi, false, bhi, (short)0, acc[nt], false, false);
      acc[nt] = __builtin_amdgcn_wmma_f32_16x16x32_f16(
          false, alo, false, bhi, (short)0, acc[nt], false, false);
      acc[nt] = __builtin_amdgcn_wmma_f32_16x16x32_f16(
          false, ahi, false, blo, (short)0, acc[nt], false, false);
    }
  }

  // ---- epilogue: add w0[k], store valid columns (C/D layout: M = r+8*half) --
  #pragma unroll
  for (int nt = 0; nt < NTILES; ++nt) {
    int kcol = nt * 16 + l16;
    if (kcol < K_OUT) {
      float w0v = w0[kcol];
      #pragma unroll
      for (int r = 0; r < 8; ++r) {
        int row = row0 + r + 8 * half;
        out[(size_t)row * K_OUT + kcol] = acc[nt][r] + w0v;
      }
    }
  }
}

extern "C" void kernel_launch(void* const* d_in, const int* in_sizes, int n_in,
                              void* d_out, int out_size, void* d_ws, size_t ws_size,
                              hipStream_t stream) {
  const float* x      = (const float*)d_in[0];
  const float* mus    = (const float*)d_in[1];
  const float* sigmas = (const float*)d_in[2];
  const float* pis    = (const float*)d_in[3];
  float* out = (float*)d_out;

  char* ws = (char*)d_ws;
  _Float16* whi = (_Float16*)ws;                                   // 28672 B
  _Float16* wlo = (_Float16*)(ws + (size_t)K_PAD * F_DIM * 2);     // 28672 B
  float*    w0  = (float*)   (ws + (size_t)2 * K_PAD * F_DIM * 2); // 448 B

  gmm_prep_kernel<<<1, 128, 0, stream>>>(mus, sigmas, pis, whi, wlo, w0);
  gmm_wmma_kernel<<<N_PIX / 128, 256, 0, stream>>>(x, whi, wlo, w0, out);
}
